// GeometryAwareAttention_84413287236162
// MI455X (gfx1250) — matrix operand order repaired and checked by reading, not compiled, not run on hardware
//
#include <hip/hip_runtime.h>
#include <cstdint>

#define B_     4
#define M_     2048
#define DMODEL 256
#define NHEAD  8
#define DK     32

typedef __attribute__((ext_vector_type(16))) _Float16 v16h;
typedef __attribute__((ext_vector_type(8)))  _Float16 v8h;
typedef __attribute__((ext_vector_type(8)))  float    v8f;

typedef __attribute__((ext_vector_type(4))) unsigned int u32x4;
typedef __attribute__((ext_vector_type(8))) int          i32x8;
typedef __attribute__((ext_vector_type(4))) int          i32x4;

#if __has_builtin(__builtin_amdgcn_tensor_load_to_lds) && __has_builtin(__builtin_amdgcn_s_wait_tensorcnt)
#define HAVE_TDM 1
#else
#define HAVE_TDM 0
#endif

static __device__ __forceinline__ v16h cat8(v8h lo, v8h hi) {
  return __builtin_shufflevector(lo, hi, 0,1,2,3,4,5,6,7,8,9,10,11,12,13,14,15);
}

#if HAVE_TDM
// Issue a TDM 2D tile load (16-bit elements) into LDS.
// Descriptor layout per CDNA5 ISA section 8 (D# groups 0/1; groups 2/3 zero for 2D).
// This toolchain exposes the 6-arg builtin:
//   (u32x4 g0, i32x8 g1, i32x4 g2, i32x4 g3, i32x8 extra, i32 cpol)
static __device__ __forceinline__ void tdm_load_2d(
    uint32_t lds_off, const void* gptr,
    uint32_t tensor_d0, uint32_t tensor_d1, uint64_t d0_stride,
    uint32_t tile_d0, uint32_t tile_d1)
{
  const uint64_t ga = (uint64_t)(uintptr_t)gptr;
  u32x4 g0;
  g0[0] = 1u;                                     // count=1 valid descriptor
  g0[1] = lds_off;                                // lds_addr (bytes)
  g0[2] = (uint32_t)ga;                           // global_addr[31:0]
  g0[3] = (uint32_t)(ga >> 32) | (2u << 30);      // global_addr[56:32] | type=2
  i32x8 g1;
  g1[0] = (int)(1u << 16);                        // workgroup_mask=0, data_size=1 (2B)
  g1[1] = (int)((tensor_d0 & 0xFFFFu) << 16);     // tensor_dim0[15:0] @ bits 63:48
  g1[2] = (int)((tensor_d0 >> 16) | ((tensor_d1 & 0xFFFFu) << 16)); // d0 hi | d1 lo
  g1[3] = (int)((tensor_d1 >> 16) | (tile_d0 << 16));               // d1 hi | tile_dim0
  g1[4] = (int)(tile_d1 & 0xFFFFu);               // tile_dim1 (tile_dim2 = 0)
  g1[5] = (int)(uint32_t)d0_stride;               // tensor_dim0_stride[31:0]
  g1[6] = (int)((uint32_t)(d0_stride >> 32) & 0xFFFFu); // stride[47:32], d1_stride=0
  g1[7] = 0;
  const i32x4 z4 = {0, 0, 0, 0};
  const i32x8 z8 = {0, 0, 0, 0, 0, 0, 0, 0};
  __builtin_amdgcn_tensor_load_to_lds(g0, g1, z4, z4, z8, 0);
}
#endif

// ---------------------------------------------------------------------------
// Kernel 1: fused QKV projection.  out[m][n] = sum_k x[m][k] * W[n][k] + b[n]
// One wave computes a 16x16 tile with 8 x V_WMMA_F32_16X16X32_F16 k-steps.
// Q is pre-scaled by 1/sqrt(DK).  Q/K layout: [B,H,M,DK] (head-vectors
// contiguous); V layout: [B,H,DK,M] (transposed, so the flash kernel's P@V
// B-fragments / TDM tiles are contiguous over keys).
// ---------------------------------------------------------------------------
__global__ __launch_bounds__(256) void qkv_kernel(
    const float* __restrict__ x,
    const float* __restrict__ wq, const float* __restrict__ bq,
    const float* __restrict__ wk, const float* __restrict__ bk,
    const float* __restrict__ wv, const float* __restrict__ bv,
    _Float16* __restrict__ qh, _Float16* __restrict__ kh, _Float16* __restrict__ vhT)
{
  const int lane = threadIdx.x & 31;
  const int wid  = blockIdx.x * 8 + (threadIdx.x >> 5);
  const int row_tile = wid / 48;              // 512 row tiles over B*M = 8192
  const int col_tile = wid - row_tile * 48;   // 48 col tiles over 3*256
  const int mat = col_tile >> 4;              // 0=Q, 1=K, 2=V
  const int n0  = (col_tile & 15) << 4;
  const int m0  = row_tile << 4;

  const float* W    = (mat == 0) ? wq : (mat == 1) ? wk : wv;
  const float* bias = (mat == 0) ? bq : (mat == 1) ? bk : bv;
  _Float16* dst     = (mat == 0) ? qh : (mat == 1) ? kh : vhT;
  const float scale = (mat == 0) ? 0.17677669529663687f /* 1/sqrt(32) */ : 1.0f;

  const int lcol = lane & 15;
  const int hi   = lane >> 4;

  // A fragment: lane<16 -> row lcol, K halves {0..7,16..23}; lane>=16 -> {8..15,24..31}
  const float* arow = x + (m0 + lcol) * DMODEL;
  // B fragment: column n = n0+lcol, K halves {hi*8..+7, 16+hi*8..+7} (W[n][k] contiguous in k)
  const float* brow = W + (n0 + lcol) * DMODEL;

  v8f c = {};
  for (int kk = 0; kk < DMODEL; kk += 32) {
    v16h a, b;
    const float* ap = arow + kk + hi * 8;
    const float* bp = brow + kk + hi * 8;
#pragma unroll
    for (int i = 0; i < 8; ++i) { a[i] = (_Float16)ap[i]; a[8 + i] = (_Float16)ap[16 + i]; }
#pragma unroll
    for (int i = 0; i < 8; ++i) { b[i] = (_Float16)bp[i]; b[8 + i] = (_Float16)bp[16 + i]; }
    c = __builtin_amdgcn_wmma_f32_16x16x32_f16(false, a, false, b, (short)0, c, false, false);
  }

  // C/D layout: VGPR v -> row (v + 8*hi), col lcol
  const int ocol = n0 + lcol;
  const int h  = ocol >> 5;
  const int dk = ocol & 31;
  const float bs = bias[ocol];
#pragma unroll
  for (int v = 0; v < 8; ++v) {
    const int orow = m0 + v + (hi << 3);         // flat b*M + m
    const int bb = orow >> 11, mm = orow & (M_ - 1);
    const _Float16 val = (_Float16)((c[v] + bs) * scale);
    if (mat == 2)
      dst[(((size_t)bb * NHEAD + h) * DK + dk) * M_ + mm] = val;   // V transposed
    else
      dst[(((size_t)bb * NHEAD + h) * M_ + mm) * DK + dk] = val;   // Q/K row-major
  }
}

// ---------------------------------------------------------------------------
// Kernel 2: flash attention with fused geometric bias.
// Block = one (b,h) x 128 query rows; 8 waves x 16 rows each.
// K/V 32-key tiles are double-buffered in LDS, staged by the Tensor Data
// Mover (wave 0 issues descriptors, block syncs via s_wait_tensorcnt +
// barrier); falls back to cooperative b64 staging if the TDM builtin is
// unavailable.  Per 32-key step: 2 WMMA (S), online softmax via 16-lane
// shfl butterflies, P transposed C->A layout through per-wave LDS, 2 WMMA (P@V).
// ---------------------------------------------------------------------------
__global__ __launch_bounds__(256) void flash_kernel(
    const _Float16* __restrict__ qh, const _Float16* __restrict__ kh,
    const _Float16* __restrict__ vhT, const float* __restrict__ coords,
    const float* __restrict__ alphap, float* __restrict__ out)
{
  __shared__ __align__(32) _Float16 sk [2][32 * DK];   // [key][dk]
  __shared__ __align__(32) _Float16 svT[2][DK * 32];   // [dk][key]
  __shared__ __align__(32) _Float16 sp [8][16 * 32];   // per-wave P [row][key]

  const int lane = threadIdx.x & 31;
  const int wave = threadIdx.x >> 5;
  const int bh   = blockIdx.x >> 4;                    // 0..31
  const int b    = bh >> 3, h = bh & 7;
  const int m0   = ((blockIdx.x & 15) << 7) + (wave << 4);

  const _Float16* Qb  = qh  + (size_t)bh * M_ * DK;
  const _Float16* Kb  = kh  + (size_t)bh * M_ * DK;   // [key][dk]
  const _Float16* VbT = vhT + (size_t)bh * DK * M_;   // [dk][key]
  const float*    Cb  = coords + (size_t)b * M_ * 3;
  const float alpha = *alphap;

  const int lcol = lane & 15;
  const int hi   = lane >> 4;

  // Q A-fragment (16 rows x DK=32), resident for the whole loop.
  v16h aq;
  {
    const _Float16* qp = Qb + (m0 + lcol) * DK + hi * 8;
    aq = cat8(*(const v8h*)qp, *(const v8h*)(qp + 16));
  }
  // Row coordinates for the bias; row for accumulator slot v is m0+v+8*hi.
  float cx[8], cy[8], cz[8];
#pragma unroll
  for (int v = 0; v < 8; ++v) {
    const float* cp = Cb + (m0 + v + (hi << 3)) * 3;
    cx[v] = cp[0]; cy[v] = cp[1]; cz[v] = cp[2];
  }

  v8f o0 = {}, o1 = {};
  float Mr[8], Lr[8];
#pragma unroll
  for (int v = 0; v < 8; ++v) { Mr[v] = -3.0e38f; Lr[v] = 0.0f; }

  const int NJB = M_ / 32;

#if HAVE_TDM
  const uint32_t sk_off[2]  = { (uint32_t)(uintptr_t)&sk[0][0],  (uint32_t)(uintptr_t)&sk[1][0]  };
  const uint32_t svT_off[2] = { (uint32_t)(uintptr_t)&svT[0][0], (uint32_t)(uintptr_t)&svT[1][0] };
  if (wave == 0) {
    // Prime both buffers: pairs (K,V) for jb=0 and jb=1.  TENSORcnt = 4.
#pragma unroll
    for (int j = 0; j < 2; ++j) {
      // K tile: innermost dim dk (32 contiguous), 32 key rows, row stride DK.
      tdm_load_2d(sk_off[j],  Kb + (size_t)j * 32 * DK, DK, M_, DK, 32, 32);
      // V tile: innermost dim keys (32 contiguous at offset jb*32), 32 dk rows, stride M.
      tdm_load_2d(svT_off[j], VbT + (size_t)j * 32,     M_, DK, M_, 32, 32);
    }
  }
#else
  // Cooperative fallback staging: 256 threads, one b64 each per tile.
  const int t  = threadIdx.x;
  const int r8 = t >> 3;          // 0..31: key row (K) / dk row (V)
  const int c4 = (t & 7) << 2;    // 0..28: 4-element column group
  *(uint64_t*)&sk[0][r8 * DK + c4]  = *(const uint64_t*)&Kb[r8 * DK + c4];
  *(uint64_t*)&svT[0][r8 * 32 + c4] = *(const uint64_t*)&VbT[r8 * M_ + c4];
#endif
  __syncthreads();

  for (int jb = 0; jb < NJB; ++jb) {
    const int buf = jb & 1;

#if HAVE_TDM
    // In-order TDM completion: <=2 outstanding means this buffer's (K,V) landed.
    if (jb + 1 < NJB) __builtin_amdgcn_s_wait_tensorcnt(2);
    else              __builtin_amdgcn_s_wait_tensorcnt(0);
    __syncthreads();
#else
    if (jb + 1 < NJB) {                       // prefetch next block into buf^1
      const int nb = buf ^ 1;
      const int j1 = (jb + 1) * 32;
      *(uint64_t*)&sk[nb][r8 * DK + c4]  = *(const uint64_t*)&Kb[(size_t)(j1 + r8) * DK + c4];
      *(uint64_t*)&svT[nb][r8 * 32 + c4] = *(const uint64_t*)&VbT[(size_t)r8 * M_ + j1 + c4];
    }
#endif

    // S = Q @ K^T for 32 keys: two 16x16 tiles.
    // B-fragment: lane -> key column lcol (+16), K(=dk) halves {hi*8..+7, 16+hi*8..+7}.
    v16h bk0 = cat8(*(const v8h*)&sk[buf][lcol * DK + hi * 8], *(const v8h*)&sk[buf][lcol * DK + hi * 8 + 16]);
    v16h bk1 = cat8(*(const v8h*)&sk[buf][(16 + lcol) * DK + hi * 8], *(const v8h*)&sk[buf][(16 + lcol) * DK + hi * 8 + 16]);
    v8f z = {};
    v8f s0 = __builtin_amdgcn_wmma_f32_16x16x32_f16(false, aq, false, bk0, (short)0, z, false, false);
    v8f s1 = __builtin_amdgcn_wmma_f32_16x16x32_f16(false, aq, false, bk1, (short)0, z, false, false);

    // Geometric bias + online softmax.
    const int j0 = jb * 32;
    const float* cj0 = Cb + (j0 + lcol) * 3;
    const float* cj1 = Cb + (j0 + 16 + lcol) * 3;
    const float kx0 = cj0[0], ky0 = cj0[1], kz0 = cj0[2];
    const float kx1 = cj1[0], ky1 = cj1[1], kz1 = cj1[2];

#pragma unroll
    for (int v = 0; v < 8; ++v) {
      float t0 = s0[v] + alpha * (cx[v] * kx0 + cy[v] * ky0 + cz[v] * kz0);
      float t1 = s1[v] + alpha * (cx[v] * kx1 + cy[v] * ky1 + cz[v] * kz1);
      // Row v+8*hi lives in lanes {hi*16 .. hi*16+15}: xor masks < 16 stay in-half.
      float mx = fmaxf(t0, t1);
#pragma unroll
      for (int off = 1; off < 16; off <<= 1) mx = fmaxf(mx, __shfl_xor(mx, off, 32));
      const float Mn   = fmaxf(Mr[v], mx);
      const float corr = __expf(Mr[v] - Mn);
      const float p0 = __expf(t0 - Mn);
      const float p1 = __expf(t1 - Mn);
      float ps = p0 + p1;
#pragma unroll
      for (int off = 1; off < 16; off <<= 1) ps += __shfl_xor(ps, off, 32);
      Lr[v] = Lr[v] * corr + ps;
      Mr[v] = Mn;
      o0[v] *= corr;
      o1[v] *= corr;
      const int prow = v + (hi << 3);
      sp[wave][prow * 32 + lcol]      = (_Float16)p0;   // C-layout -> row-major LDS
      sp[wave][prow * 32 + 16 + lcol] = (_Float16)p1;
    }

    // Reload P as an A-fragment (16 rows x 32 keys) — the LDS round trip
    // performs the C->A lane transpose; same-wave DS ops are ordered.
    v16h ap;
    {
      const _Float16* pp = &sp[wave][lcol * 32 + hi * 8];
      ap = cat8(*(const v8h*)pp, *(const v8h*)(pp + 16));
    }
    // V B-fragments from transposed tile: column dk = lcol (+16), keys contiguous.
    v16h bv0 = cat8(*(const v8h*)&svT[buf][lcol * 32 + hi * 8], *(const v8h*)&svT[buf][lcol * 32 + hi * 8 + 16]);
    v16h bv1 = cat8(*(const v8h*)&svT[buf][(16 + lcol) * 32 + hi * 8], *(const v8h*)&svT[buf][(16 + lcol) * 32 + hi * 8 + 16]);

    o0 = __builtin_amdgcn_wmma_f32_16x16x32_f16(false, ap, false, bv0, (short)0, o0, false, false);
    o1 = __builtin_amdgcn_wmma_f32_16x16x32_f16(false, ap, false, bv1, (short)0, o1, false, false);

    __syncthreads();   // all waves done with buf before it is refilled

#if HAVE_TDM
    if (wave == 0 && jb + 2 < NJB) {
      const int j2 = jb + 2;
      tdm_load_2d(sk_off[buf],  Kb + (size_t)j2 * 32 * DK, DK, M_, DK, 32, 32);
      tdm_load_2d(svT_off[buf], VbT + (size_t)j2 * 32,     M_, DK, M_, 32, 32);
    }
#endif
  }

  // Epilogue: out[b][m][h*32 + dk] = O / L
#pragma unroll
  for (int v = 0; v < 8; ++v) {
    const int r = m0 + v + (hi << 3);
    const float inv = 1.0f / Lr[v];
    float* op = out + ((size_t)b * M_ + r) * DMODEL + h * DK;
    op[lcol]      = o0[v] * inv;
    op[16 + lcol] = o1[v] * inv;
  }
}

// ---------------------------------------------------------------------------
extern "C" void kernel_launch(void* const* d_in, const int* in_sizes, int n_in,
                              void* d_out, int out_size, void* d_ws, size_t ws_size,
                              hipStream_t stream) {
  const float* x      = (const float*)d_in[0];
  const float* coords = (const float*)d_in[1];
  const float* wq     = (const float*)d_in[2];
  const float* bq     = (const float*)d_in[3];
  const float* wk     = (const float*)d_in[4];
  const float* bk     = (const float*)d_in[5];
  const float* wv     = (const float*)d_in[6];
  const float* bv     = (const float*)d_in[7];
  const float* alpha  = (const float*)d_in[8];
  float* out = (float*)d_out;

  const size_t elems = (size_t)B_ * NHEAD * M_ * DK;   // 2,097,152 per tensor
  _Float16* qh  = (_Float16*)d_ws;
  _Float16* kh  = qh + elems;
  _Float16* vhT = kh + elems;

  // 512 row tiles * 48 col tiles = 24576 wave-tiles / 8 waves = 3072 blocks
  qkv_kernel<<<3072, 256, 0, stream>>>(x, wq, bq, wk, bk, wv, bv, qh, kh, vhT);
  // 32 (b,h) pairs * 16 query slabs = 512 blocks
  flash_kernel<<<512, 256, 0, stream>>>(qh, kh, vhT, coords, alpha, out);
}
